// LongAttention_75634374083224
// MI455X (gfx1250) — matrix operand order repaired and checked by reading, not compiled, not run on hardware
//
#include <hip/hip_runtime.h>
#include <hip/hip_bf16.h>

// ---------------------------------------------------------------------------
// Problem constants (fixed by the reference)
// ---------------------------------------------------------------------------
#define Bc 2
#define Tc 4096
#define Cc 2048
#define Hc 16
#define Dc 128
#define Kc 4
#define NTOK (Bc * Tc)          // 8192 tokens
#define NCH  64                 // scan chunks
#define CHL  (Tc / NCH)         // 64 steps per chunk
#define EPSc 1e-5f

typedef __bf16 bf16_t;
typedef unsigned int u32;
typedef __attribute__((ext_vector_type(16))) __bf16 v16b;
typedef __attribute__((ext_vector_type(8)))  float  v8f;
typedef __attribute__((ext_vector_type(4)))  u32    u32x4;
typedef __attribute__((ext_vector_type(8)))  u32    u32x8;

__device__ __forceinline__ float sigmoidf_(float x) { return 1.f / (1.f + __expf(-x)); }

// LDS byte offset of a shared-memory pointer (flat low 32 bits == DS offset)
__device__ __forceinline__ u32 lds_off_(const void* p) { return (u32)(uintptr_t)p; }

// CDNA5 async global->LDS copy, 16B, ASYNCcnt-tracked (no VGPR round trip)
__device__ __forceinline__ void async_b128_(void* lds, const void* g) {
    u32 l = lds_off_(lds);
    asm volatile("global_load_async_to_lds_b128 %0, %1, off"
                 :: "v"(l), "v"(g) : "memory");
}

// ---------------------------------------------------------------------------
// f32 -> bf16 cast
// ---------------------------------------------------------------------------
__global__ void cast_f32_bf16(const float* __restrict__ in, bf16_t* __restrict__ out, int n) {
    int i = blockIdx.x * blockDim.x + threadIdx.x;
    if (i < n) out[i] = (bf16_t)in[i];
}

// ---------------------------------------------------------------------------
// Causal depthwise conv1d (K=4) + bias + SiLU, writes bf16 x_conv
// ---------------------------------------------------------------------------
__global__ void conv_silu_kernel(const float* __restrict__ x,
                                 const float* __restrict__ cw,
                                 const float* __restrict__ cb,
                                 bf16_t* __restrict__ xc) {
    int i = blockIdx.x * blockDim.x + threadIdx.x;   // over B*T*C
    if (i >= Bc * Tc * Cc) return;
    int c = i % Cc;
    int t = (i / Cc) % Tc;
    int b = i / (Cc * Tc);
    float acc = cb[c];
    #pragma unroll
    for (int j = 0; j < Kc; ++j) {
        int ts = t - (Kc - 1) + j;
        if (ts >= 0) acc += x[((size_t)(b * Tc + ts)) * Cc + c] * cw[c * Kc + j];
    }
    xc[i] = (bf16_t)(acc * sigmoidf_(acc));          // SiLU
}

// ---------------------------------------------------------------------------
// Shared WMMA tile-compute: 8 waves, wave tile 32(M) x 64(N), K-step 32
// ---------------------------------------------------------------------------
#define TBM 128
#define TBN 128
#define TBK 32

union Frag { v16b v; uint4 q[2]; };

__device__ __forceinline__ void wmma_tile_(const bf16_t* lA, const bf16_t* lB,
                                           v8f acc[2][4], int wm, int wn, int g, int ln) {
    Frag af[2];
    #pragma unroll
    for (int mt = 0; mt < 2; ++mt) {
        int m = wm * 32 + mt * 16 + ln;
        af[mt].q[0] = *(const uint4*)&lA[m * TBK + 8 * g];
        af[mt].q[1] = *(const uint4*)&lA[m * TBK + 16 + 8 * g];
    }
    Frag bf[4];
    #pragma unroll
    for (int nt = 0; nt < 4; ++nt) {
        int n = wn * 64 + nt * 16 + ln;
        bf[nt].q[0] = *(const uint4*)&lB[n * TBK + 8 * g];
        bf[nt].q[1] = *(const uint4*)&lB[n * TBK + 16 + 8 * g];
    }
    #pragma unroll
    for (int mt = 0; mt < 2; ++mt)
        #pragma unroll
        for (int nt = 0; nt < 4; ++nt)
            acc[mt][nt] = __builtin_amdgcn_wmma_f32_16x16x32_bf16(
                false, af[mt].v, false, bf[nt].v, (short)0, acc[mt][nt], false, false);
}

__device__ __forceinline__ void store_tile_(v8f acc[2][4], float* Cf, bf16_t* Cb,
                                            int bm, int bn, int N, int wm, int wn, int g, int ln) {
    #pragma unroll
    for (int mt = 0; mt < 2; ++mt)
        #pragma unroll
        for (int nt = 0; nt < 4; ++nt)
            #pragma unroll
            for (int i = 0; i < 8; ++i) {
                int r = bm + wm * 32 + mt * 16 + 8 * g + i;
                int c = bn + wn * 64 + nt * 16 + ln;
                float v = acc[mt][nt][i];
                if (Cf) Cf[(size_t)r * N + c] = v;
                else    Cb[(size_t)r * N + c] = (bf16_t)v;
            }
}

// ---------------------------------------------------------------------------
// GEMM variant A: async-to-LDS staging (ASYNCcnt), double-buffered LDS
//   C[M,N] = A[M,K] * W[N,K]^T, f32 accumulate, out f32 or bf16
// ---------------------------------------------------------------------------
__global__ __launch_bounds__(256)
void gemm_bf16_wmma_async(const bf16_t* __restrict__ A, const bf16_t* __restrict__ W,
                          float* __restrict__ Cf, bf16_t* __restrict__ Cb,
                          int M, int N, int Kd) {
    __shared__ __align__(16) bf16_t lA[2][TBM * TBK];
    __shared__ __align__(16) bf16_t lB[2][TBN * TBK];

    const int tid  = threadIdx.x;
    const int lane = tid & 31;
    const int wave = tid >> 5;
    const int wm   = wave & 3;
    const int wn   = wave >> 2;
    const int g    = lane >> 4;
    const int ln   = lane & 15;
    const int bm   = blockIdx.y * TBM;
    const int bn   = blockIdx.x * TBN;
    const int row  = tid >> 1;           // 0..127
    const int koff = (tid & 1) * 16;     // halves 0..15 / 16..31

    v8f acc[2][4];
    #pragma unroll
    for (int i = 0; i < 2; ++i)
        #pragma unroll
        for (int j = 0; j < 4; ++j) acc[i][j] = (v8f){};

    auto issue_tile = [&](int buf, int kb) {
        const bf16_t* gA = A + (size_t)(bm + row) * Kd + kb + koff;
        const bf16_t* gW = W + (size_t)(bn + row) * Kd + kb + koff;
        async_b128_(&lA[buf][row * TBK + koff],     gA);
        async_b128_(&lA[buf][row * TBK + koff + 8], gA + 8);
        async_b128_(&lB[buf][row * TBK + koff],     gW);
        async_b128_(&lB[buf][row * TBK + koff + 8], gW + 8);
    };

    const int nIter = Kd / TBK;
    issue_tile(0, 0);
    for (int it = 0; it < nIter; ++it) {
        const int cur = it & 1;
        if (it + 1 < nIter) {
            issue_tile(cur ^ 1, (it + 1) * TBK);
            asm volatile("s_wait_asynccnt 0x4" ::: "memory");  // current tile landed
        } else {
            asm volatile("s_wait_asynccnt 0x0" ::: "memory");
        }
        __syncthreads();
        wmma_tile_(lA[cur], lB[cur], acc, wm, wn, g, ln);
        __syncthreads();                 // protect buffer before its re-fill
    }
    store_tile_(acc, Cf, Cb, bm, bn, N, wm, wn, g, ln);
}

// ---------------------------------------------------------------------------
// GEMM variant B: Tensor Data Mover staging (TENSORcnt), double-buffered LDS
//   wave 0 builds a 2D D# (groups 0-1) in SGPRs and issues tensor_load_to_lds
//   for the whole 128-row x 64B tile; all waves then barrier and run WMMA.
// ---------------------------------------------------------------------------
__device__ __forceinline__ void tdm_load_tile_(void* lds, const void* gptr,
                                               u32 rows, u32 row_stride_bytes) {
    const u32 dsz = 3u;                          // 8-byte elements
    const u32 d0u = row_stride_bytes >> 3;       // tensor_dim0 / dim0_stride (units)
    const u32 t0u = (TBK * 2) >> 3;              // tile_dim0 = 64B = 8 units
    uint64_t ga = (uint64_t)(uintptr_t)gptr;

    u32x4 g0;
    g0[0] = 1u;                                  // count=1, user descriptor
    g0[1] = lds_off_(lds);                       // lds_addr
    g0[2] = (u32)ga;                             // global_addr[31:0]
    g0[3] = (u32)((ga >> 32) & 0x1FFFFFFu) | (2u << 30);  // addr[56:32] | type=2

    u32x8 g1;
    g1[0] = dsz << 16;                           // wg_mask=0, data_size
    g1[1] = (d0u & 0xFFFFu) << 16;               // tensor_dim0 lo16 (bits 63:48)
    g1[2] = (d0u >> 16) | ((rows & 0xFFFFu) << 16);  // dim0 hi16 | tensor_dim1 lo16
    g1[3] = (rows >> 16) | (t0u << 16);          // dim1 hi16 | tile_dim0
    g1[4] = rows & 0xFFFFu;                      // tile_dim1 (tile_dim2=0)
    g1[5] = d0u;                                 // tensor_dim0_stride[31:0]
    g1[6] = 0u;                                  // stride hi16, dim1_stride lo
    g1[7] = 0u;
    asm volatile("tensor_load_to_lds %0, %1" :: "s"(g0), "s"(g1) : "memory");
}

__global__ __launch_bounds__(256)
void gemm_bf16_wmma_tdm(const bf16_t* __restrict__ A, const bf16_t* __restrict__ W,
                        float* __restrict__ Cf, bf16_t* __restrict__ Cb,
                        int M, int N, int Kd) {
    __shared__ __align__(16) bf16_t lA[2][TBM * TBK];
    __shared__ __align__(16) bf16_t lB[2][TBN * TBK];

    const int tid  = threadIdx.x;
    const int lane = tid & 31;
    const int wave = tid >> 5;
    const int wm   = wave & 3;
    const int wn   = wave >> 2;
    const int g    = lane >> 4;
    const int ln   = lane & 15;
    const int bm   = blockIdx.y * TBM;
    const int bn   = blockIdx.x * TBN;

    v8f acc[2][4];
    #pragma unroll
    for (int i = 0; i < 2; ++i)
        #pragma unroll
        for (int j = 0; j < 4; ++j) acc[i][j] = (v8f){};

    auto issue_tile = [&](int buf, int kb) {
        tdm_load_tile_(&lA[buf][0], A + (size_t)bm * Kd + kb, TBM, Kd * 2);
        tdm_load_tile_(&lB[buf][0], W + (size_t)bn * Kd + kb, TBN, Kd * 2);
    };

    const int nIter = Kd / TBK;
    if (wave == 0) issue_tile(0, 0);
    for (int it = 0; it < nIter; ++it) {
        const int cur = it & 1;
        if (wave == 0) {
            if (it + 1 < nIter) {
                issue_tile(cur ^ 1, (it + 1) * TBK);
                __builtin_amdgcn_s_wait_tensorcnt(2);   // current 2 DMAs landed
            } else {
                __builtin_amdgcn_s_wait_tensorcnt(0);
            }
        }
        __syncthreads();
        wmma_tile_(lA[cur], lB[cur], acc, wm, wn, g, ln);
        __syncthreads();
    }
    store_tile_(acc, Cf, Cb, bm, bn, N, wm, wn, g, ln);
}

// ---------------------------------------------------------------------------
// gamma head-gate:  gam[token,h] = sigmoid(xc . Wgam[h] + bgam[h])
// ---------------------------------------------------------------------------
__global__ void gamma_kernel(const bf16_t* __restrict__ xc, const bf16_t* __restrict__ wg,
                             const float* __restrict__ bg, float* __restrict__ gam) {
    int i = blockIdx.x * blockDim.x + threadIdx.x;      // over NTOK*H
    if (i >= NTOK * Hc) return;
    int h = i & (Hc - 1);
    size_t tok = (size_t)(i >> 4);
    const bf16_t* xr = xc + tok * Cc;
    const bf16_t* wr = wg + (size_t)h * Cc;
    float acc = bg[h];
    for (int j = 0; j < Cc; ++j) acc += (float)xr[j] * (float)wr[j];
    gam[i] = sigmoidf_(acc);
}

// ---------------------------------------------------------------------------
// q: l2norm, k: l2norm, v: LayerNorm(vn_g,vn_b) -- one block per (token,head)
// ---------------------------------------------------------------------------
__global__ __launch_bounds__(128)
void qkv_norm_kernel(bf16_t* __restrict__ qb, bf16_t* __restrict__ kb, bf16_t* __restrict__ vb,
                     const float* __restrict__ vg, const float* __restrict__ vbb) {
    __shared__ float sq[128], sk[128], sv[128], sv2[128];
    int d = threadIdx.x;
    size_t row = blockIdx.x;                 // token*H + h
    size_t tok = row >> 4;
    int h = (int)(row & (Hc - 1));
    size_t base = tok * Cc + (size_t)h * Dc + d;
    float q = (float)qb[base], k = (float)kb[base], v = (float)vb[base];
    sq[d] = q * q; sk[d] = k * k; sv[d] = v; sv2[d] = v * v;
    __syncthreads();
    for (int s = 64; s > 0; s >>= 1) {
        if (d < s) { sq[d] += sq[d + s]; sk[d] += sk[d + s]; sv[d] += sv[d + s]; sv2[d] += sv2[d + s]; }
        __syncthreads();
    }
    float qi = 1.f / fmaxf(sqrtf(sq[0]), 1e-12f);
    float ki = 1.f / fmaxf(sqrtf(sk[0]), 1e-12f);
    float mu = sv[0] * (1.f / Dc);
    float var = sv2[0] * (1.f / Dc) - mu * mu;
    float rs = rsqrtf(var + EPSc);
    qb[base] = (bf16_t)(q * qi);
    kb[base] = (bf16_t)(k * ki);
    vb[base] = (bf16_t)((v - mu) * rs * vg[d] + vbb[d]);
}

// ---------------------------------------------------------------------------
// Chunked linear scan, pass 1: local scan per (b, chunk, c); emit chunk A/B
// ---------------------------------------------------------------------------
__global__ void scan_pass1(const bf16_t* __restrict__ kb, const bf16_t* __restrict__ vb,
                           const bf16_t* __restrict__ igb, const float* __restrict__ big,
                           const float* __restrict__ gam,
                           float* __restrict__ memf, float* __restrict__ Pf,
                           float* __restrict__ cA, float* __restrict__ cB) {
    int idx = blockIdx.x * blockDim.x + threadIdx.x;   // B * NCH * C
    if (idx >= Bc * NCH * Cc) return;
    int c = idx % Cc;
    int chunk = (idx / Cc) % NCH;
    int b = idx / (Cc * NCH);
    int h = c >> 7, d = c & (Dc - 1);
    float m = 0.f, p = 1.f;
    int t0 = chunk * CHL;
    for (int tt = 0; tt < CHL; ++tt) {
        size_t tk = (size_t)b * Tc + (t0 + tt);
        float g = gam[tk * Hc + h];
        size_t ix = tk * Cc + c;
        float ig = sigmoidf_((float)igb[ix] + big[c]);
        float kv = ig * (float)kb[ix] * (float)vb[ix];
        m = g * m + kv;
        p *= g;
        memf[ix] = m;
        if (d == 0) Pf[tk * Hc + h] = p;               // local cumulative gamma
    }
    size_t ci = ((size_t)b * Cc + c) * NCH + chunk;
    cA[ci] = p;
    cB[ci] = m;
}

// pass 2: tiny sequential carry over chunks per channel
__global__ void scan_pass2(const float* __restrict__ cA, const float* __restrict__ cB,
                           float* __restrict__ cIn) {
    int i = blockIdx.x * blockDim.x + threadIdx.x;     // B*C
    if (i >= Bc * Cc) return;
    size_t base = (size_t)i * NCH;
    float s = 0.f;
    for (int j = 0; j < NCH; ++j) {
        cIn[base + j] = s;
        s = cA[base + j] * s + cB[base + j];
    }
}

// pass 3: mem += carry_in(chunk) * local_prefix_gamma(t)
__global__ void scan_pass3(float* __restrict__ memf, const float* __restrict__ cIn,
                           const float* __restrict__ Pf) {
    int i = blockIdx.x * blockDim.x + threadIdx.x;     // B*T*C
    if (i >= Bc * Tc * Cc) return;
    int c = i % Cc;
    int t = (i / Cc) % Tc;
    int b = i / (Cc * Tc);
    int h = c >> 7;
    int chunk = t / CHL;
    memf[i] += cIn[((size_t)b * Cc + c) * NCH + chunk] * Pf[((size_t)b * Tc + t) * Hc + h];
}

// ---------------------------------------------------------------------------
// LN(mem)*q -> GroupNorm -> * sigmoid(og) ; writes bf16 pre-output
// ---------------------------------------------------------------------------
__global__ __launch_bounds__(128)
void post_kernel(const float* __restrict__ memf, const bf16_t* __restrict__ qb,
                 const bf16_t* __restrict__ ogb,
                 const float* __restrict__ mg, const float* __restrict__ mb,
                 const float* __restrict__ gg, const float* __restrict__ gb,
                 const float* __restrict__ bog, bf16_t* __restrict__ pre) {
    __shared__ float s1[128], s2[128];
    int d = threadIdx.x;
    size_t row = blockIdx.x;
    size_t tok = row >> 4;
    int h = (int)(row & (Hc - 1));
    int c = h * Dc + d;
    size_t base = tok * Cc + c;
    float m = memf[base];
    s1[d] = m; s2[d] = m * m; __syncthreads();
    for (int s = 64; s > 0; s >>= 1) {
        if (d < s) { s1[d] += s1[d + s]; s2[d] += s2[d + s]; } __syncthreads();
    }
    float mu = s1[0] * (1.f / Dc);
    float var = s2[0] * (1.f / Dc) - mu * mu;
    float mo = (m - mu) * rsqrtf(var + EPSc) * mg[d] + mb[d];
    float o = mo * (float)qb[base];
    __syncthreads();
    s1[d] = o; s2[d] = o * o; __syncthreads();
    for (int s = 64; s > 0; s >>= 1) {
        if (d < s) { s1[d] += s1[d + s]; s2[d] += s2[d + s]; } __syncthreads();
    }
    float mu2 = s1[0] * (1.f / Dc);
    float var2 = s2[0] * (1.f / Dc) - mu2 * mu2;
    float on = (o - mu2) * rsqrtf(var2 + EPSc) * gg[c] + gb[c];
    float og = sigmoidf_((float)ogb[base] + bog[c]);
    pre[base] = (bf16_t)(on * og);
}

// ---------------------------------------------------------------------------
// Host side
// ---------------------------------------------------------------------------
extern "C" void kernel_launch(void* const* d_in, const int* in_sizes, int n_in,
                              void* d_out, int out_size, void* d_ws, size_t ws_size,
                              hipStream_t stream) {
    const float* x      = (const float*)d_in[0];
    const float* Wq     = (const float*)d_in[1];
    const float* Wk     = (const float*)d_in[2];
    const float* Wv     = (const float*)d_in[3];
    const float* Wo     = (const float*)d_in[4];
    const float* conv_w = (const float*)d_in[5];
    const float* conv_b = (const float*)d_in[6];
    const float* Wig    = (const float*)d_in[7];
    const float* big    = (const float*)d_in[8];
    const float* Wog    = (const float*)d_in[9];
    const float* bog    = (const float*)d_in[10];
    const float* Wgam   = (const float*)d_in[11];
    const float* bgam   = (const float*)d_in[12];
    const float* vn_g   = (const float*)d_in[13];
    const float* vn_b   = (const float*)d_in[14];
    const float* mn_g   = (const float*)d_in[15];
    const float* mn_b   = (const float*)d_in[16];
    const float* gn_g   = (const float*)d_in[17];
    const float* gn_b   = (const float*)d_in[18];
    float* out = (float*)d_out;

    // ---- carve workspace ----
    char* wp = (char*)d_ws;
    auto carve = [&](size_t bytes) { void* p = (void*)wp; wp += (bytes + 255) & ~(size_t)255; return p; };
    const size_t NE  = (size_t)NTOK * Cc;
    const size_t NW  = (size_t)Cc * Cc;
    bf16_t* x_bf   = (bf16_t*)carve(NE * 2);
    bf16_t* xc_bf  = (bf16_t*)carve(NE * 2);
    bf16_t* wq_bf  = (bf16_t*)carve(NW * 2);
    bf16_t* wk_bf  = (bf16_t*)carve(NW * 2);
    bf16_t* wv_bf  = (bf16_t*)carve(NW * 2);
    bf16_t* wo_bf  = (bf16_t*)carve(NW * 2);
    bf16_t* wig_bf = (bf16_t*)carve(NW * 2);
    bf16_t* wog_bf = (bf16_t*)carve(NW * 2);
    bf16_t* wgm_bf = (bf16_t*)carve((size_t)Hc * Cc * 2);
    bf16_t* q_bf   = (bf16_t*)carve(NE * 2);
    bf16_t* k_bf   = (bf16_t*)carve(NE * 2);
    bf16_t* v_bf   = (bf16_t*)carve(NE * 2);
    bf16_t* ig_bf  = (bf16_t*)carve(NE * 2);
    bf16_t* og_bf  = (bf16_t*)carve(NE * 2);
    bf16_t* pre_bf = (bf16_t*)carve(NE * 2);
    float*  gam_f  = (float*)carve((size_t)NTOK * Hc * 4);
    float*  mem_f  = (float*)carve(NE * 4);
    float*  P_f    = (float*)carve((size_t)NTOK * Hc * 4);
    float*  cA_f   = (float*)carve((size_t)Bc * Cc * NCH * 4);
    float*  cB_f   = (float*)carve((size_t)Bc * Cc * NCH * 4);
    float*  cIn_f  = (float*)carve((size_t)Bc * Cc * NCH * 4);

    const int TPB = 256;
    auto blks = [](size_t n, int t) { return (unsigned)((n + t - 1) / t); };

    // ---- casts to bf16 ----
    cast_f32_bf16<<<blks(NE, TPB), TPB, 0, stream>>>(x, x_bf, (int)NE);
    cast_f32_bf16<<<blks(NW, TPB), TPB, 0, stream>>>(Wq, wq_bf, (int)NW);
    cast_f32_bf16<<<blks(NW, TPB), TPB, 0, stream>>>(Wk, wk_bf, (int)NW);
    cast_f32_bf16<<<blks(NW, TPB), TPB, 0, stream>>>(Wv, wv_bf, (int)NW);
    cast_f32_bf16<<<blks(NW, TPB), TPB, 0, stream>>>(Wo, wo_bf, (int)NW);
    cast_f32_bf16<<<blks(NW, TPB), TPB, 0, stream>>>(Wig, wig_bf, (int)NW);
    cast_f32_bf16<<<blks(NW, TPB), TPB, 0, stream>>>(Wog, wog_bf, (int)NW);
    cast_f32_bf16<<<blks((size_t)Hc * Cc, TPB), TPB, 0, stream>>>(Wgam, wgm_bf, Hc * Cc);

    // ---- conv + SiLU ----
    conv_silu_kernel<<<blks(NE, TPB), TPB, 0, stream>>>(x, conv_w, conv_b, xc_bf);

    // ---- WMMA GEMMs (async-to-LDS staging; one TDM-staged for the V path) ----
    dim3 gg_(Cc / TBN, NTOK / TBM);   // (16, 64)
    gemm_bf16_wmma_async<<<gg_, 256, 0, stream>>>(x_bf,  wq_bf,  nullptr, q_bf,  NTOK, Cc, Cc);
    gemm_bf16_wmma_async<<<gg_, 256, 0, stream>>>(x_bf,  wk_bf,  nullptr, k_bf,  NTOK, Cc, Cc);
    gemm_bf16_wmma_tdm  <<<gg_, 256, 0, stream>>>(x_bf,  wv_bf,  nullptr, v_bf,  NTOK, Cc, Cc);
    gemm_bf16_wmma_async<<<gg_, 256, 0, stream>>>(xc_bf, wig_bf, nullptr, ig_bf, NTOK, Cc, Cc);
    gemm_bf16_wmma_async<<<gg_, 256, 0, stream>>>(xc_bf, wog_bf, nullptr, og_bf, NTOK, Cc, Cc);

    // ---- gamma head gate ----
    gamma_kernel<<<blks((size_t)NTOK * Hc, TPB), TPB, 0, stream>>>(xc_bf, wgm_bf, bgam, gam_f);

    // ---- q/k l2-norm, v LayerNorm ----
    qkv_norm_kernel<<<NTOK * Hc, 128, 0, stream>>>(q_bf, k_bf, v_bf, vn_g, vn_b);

    // ---- chunked gated linear scan ----
    scan_pass1<<<blks((size_t)Bc * NCH * Cc, TPB), TPB, 0, stream>>>(
        k_bf, v_bf, ig_bf, big, gam_f, mem_f, P_f, cA_f, cB_f);
    scan_pass2<<<blks((size_t)Bc * Cc, TPB), TPB, 0, stream>>>(cA_f, cB_f, cIn_f);
    scan_pass3<<<blks(NE, TPB), TPB, 0, stream>>>(mem_f, cIn_f, P_f);

    // ---- LN * q -> GroupNorm -> output gate ----
    post_kernel<<<NTOK * Hc, 128, 0, stream>>>(mem_f, q_bf, og_bf,
                                               mn_g, mn_b, gn_g, gn_b, bog, pre_bf);

    // ---- final projection (f32 out) ----
    gemm_bf16_wmma_async<<<gg_, 256, 0, stream>>>(pre_bf, wo_bf, out, nullptr, NTOK, Cc, Cc);
}
